// MixLlamaMLP_3659312136255
// MI455X (gfx1250) — matrix operand order repaired and checked by reading, not compiled, not run on hardware
//
#include <hip/hip_runtime.h>
#include <stdint.h>

#define D_MODEL 4096
#define D_FF    11008
#define TOKENS  8192   // B*S = 4*2048
#define LDAP    40     // padded LDS row stride (elements): 80B = 20 banks -> conflict-free

typedef __attribute__((ext_vector_type(8)))  float  v8f;
typedef __attribute__((ext_vector_type(16))) __bf16 v16bf;
typedef __attribute__((ext_vector_type(8)))  __bf16 v8bf;
typedef __attribute__((ext_vector_type(4)))  __bf16 v4bf;
typedef __attribute__((ext_vector_type(2)))  __bf16 v2bf;

#if defined(__AMDGCN__) && __has_builtin(__builtin_amdgcn_tensor_load_to_lds)
#define HAVE_TDM 1
typedef __attribute__((ext_vector_type(4))) unsigned int u32x4;
typedef __attribute__((ext_vector_type(8))) int          i32x8;
typedef __attribute__((ext_vector_type(4))) int          i32x4;
#else
#define HAVE_TDM 0
#endif

// packed fp32 -> bf16 convert; prefer HW builtin, else compiler fptrunc
// (clang lowers to v_cvt_pk_bf16_f32 on targets with bf16 cvt instructions)
static __device__ __forceinline__ v2bf cvt2(float a, float b) {
#if defined(__AMDGCN__) && __has_builtin(__builtin_amdgcn_cvt_pk_bf16_f32)
    return __builtin_amdgcn_cvt_pk_bf16_f32(a, b);
#else
    v2bf r; r[0] = (__bf16)a; r[1] = (__bf16)b; return r;
#endif
}

static __device__ __forceinline__ v4bf pack4(float4 v) {
    const v2bf lo = cvt2(v.x, v.y);
    const v2bf hi = cvt2(v.z, v.w);
    return __builtin_shufflevector(lo, hi, 0, 1, 2, 3);
}

static __device__ __forceinline__ v16bf cat8(v8bf lo, v8bf hi) {
    return __builtin_shufflevector(lo, hi, 0,1,2,3,4,5,6,7,8,9,10,11,12,13,14,15);
}

// ---------------------------------------------------------------------------
// Kernel 1: H[t,f] = silu(x . Wg[f,:]) * (x . Wu[f,:]),  H stored bf16
// Block tile 128x128, BK=32, 8 waves (wave tile 32x64), double-buffered LDS,
// global->register prefetch of tile k+1 overlapped with WMMAs on tile k.
// ---------------------------------------------------------------------------
__global__ void __launch_bounds__(256)
fused_gateup_kernel(const float* __restrict__ x,
                    const float* __restrict__ Wg,
                    const float* __restrict__ Wu,
                    unsigned short* __restrict__ Hraw)
{
    __bf16* __restrict__ H = reinterpret_cast<__bf16*>(Hraw);

    __shared__ __align__(16) __bf16 As [2][128 * LDAP];
    __shared__ __align__(16) __bf16 BsG[2][128 * LDAP];
    __shared__ __align__(16) __bf16 BsU[2][128 * LDAP];

    const int tid   = threadIdx.x;
    const int lane  = tid & 31;
    const int wave  = tid >> 5;
    const int waveM = wave & 3;    // 0..3  -> 32-row strip
    const int waveN = wave >> 2;   // 0..1  -> 64-col strip
    const int n16   = lane & 15;
    const int khalf = lane >> 4;   // 0 or 1

    const int rowBase = blockIdx.x * 128;   // token rows
    const int colBase = blockIdx.y * 128;   // d_ff cols

    v8f accG[2][4], accU[2][4];
    const v8f zacc = {0.f,0.f,0.f,0.f,0.f,0.f,0.f,0.f};
    #pragma unroll
    for (int mi = 0; mi < 2; ++mi)
        #pragma unroll
        for (int ni = 0; ni < 4; ++ni) { accG[mi][ni] = zacc; accU[mi][ni] = zacc; }

    float4 ra[4], rg[4], ru[4];             // staging registers for next K-slab

    auto loadTile = [&](int k0) {
        #pragma unroll
        for (int i = 0; i < 4; ++i) {
            const int idx = tid + i * 256;  // 0..1023 float4 slots
            const int r   = idx >> 3;       // tile row 0..127
            const int c4  = idx & 7;        // float4 within 32-wide K slab
            ra[i] = *(const float4*)(x  + (size_t)(rowBase + r) * D_MODEL + k0 + c4 * 4);
            rg[i] = *(const float4*)(Wg + (size_t)(colBase + r) * D_MODEL + k0 + c4 * 4);
            ru[i] = *(const float4*)(Wu + (size_t)(colBase + r) * D_MODEL + k0 + c4 * 4);
        }
    };
    auto storeTile = [&](int p) {
        #pragma unroll
        for (int i = 0; i < 4; ++i) {
            const int idx = tid + i * 256;
            const int r   = idx >> 3;
            const int c4  = idx & 7;
            *(v4bf*)&As [p][r * LDAP + c4 * 4] = pack4(ra[i]);
            *(v4bf*)&BsG[p][r * LDAP + c4 * 4] = pack4(rg[i]);
            *(v4bf*)&BsU[p][r * LDAP + c4 * 4] = pack4(ru[i]);
        }
    };

    loadTile(0);
    int p = 0;
    for (int k0 = 0; k0 < D_MODEL; k0 += 32, p ^= 1) {
        storeTile(p);
        __syncthreads();
        if (k0 + 32 < D_MODEL) loadTile(k0 + 32);   // overlap with WMMAs below

        v16bf aF[2];
        #pragma unroll
        for (int mi = 0; mi < 2; ++mi) {
            const int m = waveM * 32 + mi * 16 + n16;
            const v8bf lo = *(const v8bf*)&As[p][m * LDAP + khalf * 8];
            const v8bf hi = *(const v8bf*)&As[p][m * LDAP + khalf * 8 + 16];
            aF[mi] = cat8(lo, hi);
        }
        #pragma unroll
        for (int ni = 0; ni < 4; ++ni) {
            const int n = waveN * 64 + ni * 16 + n16;
            const v8bf gl = *(const v8bf*)&BsG[p][n * LDAP + khalf * 8];
            const v8bf gh = *(const v8bf*)&BsG[p][n * LDAP + khalf * 8 + 16];
            const v16bf bG = cat8(gl, gh);
            const v8bf ul = *(const v8bf*)&BsU[p][n * LDAP + khalf * 8];
            const v8bf uh = *(const v8bf*)&BsU[p][n * LDAP + khalf * 8 + 16];
            const v16bf bU = cat8(ul, uh);
            #pragma unroll
            for (int mi = 0; mi < 2; ++mi) {
                accG[mi][ni] = __builtin_amdgcn_wmma_f32_16x16x32_bf16(
                    false, aF[mi], false, bG, (short)0, accG[mi][ni], false, false);
                accU[mi][ni] = __builtin_amdgcn_wmma_f32_16x16x32_bf16(
                    false, aF[mi], false, bU, (short)0, accU[mi][ni], false, false);
            }
        }
    }

    // epilogue: silu(g)*u, store bf16 H
    #pragma unroll
    for (int mi = 0; mi < 2; ++mi) {
        #pragma unroll
        for (int ni = 0; ni < 4; ++ni) {
            #pragma unroll
            for (int r = 0; r < 8; ++r) {
                const float g = accG[mi][ni][r];
                const float u = accU[mi][ni][r];
                const float h = (g / (1.0f + __expf(-g))) * u;   // silu(g)*u
                const int row = rowBase + waveM * 32 + mi * 16 + r + 8 * khalf;
                const int col = colBase + waveN * 64 + ni * 16 + n16;
                H[(size_t)row * D_FF + col] = cvt2(h, h)[0];
            }
        }
    }
}

// ---------------------------------------------------------------------------
// Kernel 2: y[t,m] = H[t,:] . Wd[m,:]   (H bf16 via TDM, Wd fp32->bf16, y fp32)
// ---------------------------------------------------------------------------
#if HAVE_TDM
// One TDM DMA moves the whole 128x32 bf16 H tile into LDS with the padded
// 40-element row stride (pad_interval=3: every 64B row; pad_amount=3: +16B).
static __device__ __forceinline__ void tdm_load_H_tile(const __bf16* gptr, __bf16* lptr) {
    const uint64_t ga  = (uint64_t)(uintptr_t)gptr;
    const uint32_t lds = (uint32_t)(uintptr_t)lptr;

    u32x4 g0;
    g0[0] = 1u;                                                  // count=1 (valid user D#)
    g0[1] = lds;                                                 // lds_addr
    g0[2] = (uint32_t)(ga & 0xFFFFFFFFu);                        // global_addr[31:0]
    g0[3] = (uint32_t)((ga >> 32) & 0x01FFFFFFu) | (2u << 30);   // global_addr[56:32] | type=2

    i32x8 g1;
    g1[0] = (int)((1u << 16)      // data_size = 2 bytes
                | (1u << 20)      // pad_enable
                | (3u << 22)      // pad_interval: every 16 DWORDs (64B row)
                | (3u << 25));    // pad_amount: 4 DWORDs (16B)
    g1[1] = (int)((uint32_t)D_FF << 16);     // tensor_dim0[15:0] @ [63:48]
    g1[2] = (int)((uint32_t)TOKENS << 16);   // tensor_dim1[15:0] @ [95:80]
    g1[3] = (int)(32u << 16);                // tile_dim0 = 32    @ [127:112]
    g1[4] = (int)128u;                       // tile_dim1 = 128   @ [143:128]
    g1[5] = (int)D_FF;                       // tensor_dim0_stride[31:0]
    g1[6] = 0;
    g1[7] = 0;

    const i32x4 z4 = {0, 0, 0, 0};
#if __clang_major__ >= 23
    const i32x8 z8 = {0, 0, 0, 0, 0, 0, 0, 0};
    __builtin_amdgcn_tensor_load_to_lds(g0, g1, z4, z4, z8, 0);
#else
    __builtin_amdgcn_tensor_load_to_lds(g0, g1, z4, z4, 0);
#endif
}
#endif

__global__ void __launch_bounds__(256)
down_proj_kernel(const unsigned short* __restrict__ Hraw,
                 const float* __restrict__ Wd,
                 float* __restrict__ out)
{
    const __bf16* __restrict__ H = reinterpret_cast<const __bf16*>(Hraw);

    __shared__ __align__(16) __bf16 As[2][128 * LDAP];
    __shared__ __align__(16) __bf16 Bs[2][128 * LDAP];

    const int tid   = threadIdx.x;
    const int lane  = tid & 31;
    const int wave  = tid >> 5;
    const int waveM = wave & 3;
    const int waveN = wave >> 2;
    const int n16   = lane & 15;
    const int khalf = lane >> 4;

    const int rowBase = blockIdx.x * 128;   // token rows
    const int colBase = blockIdx.y * 128;   // d_model cols

    v8f acc[2][4];
    const v8f zacc = {0.f,0.f,0.f,0.f,0.f,0.f,0.f,0.f};
    #pragma unroll
    for (int mi = 0; mi < 2; ++mi)
        #pragma unroll
        for (int ni = 0; ni < 4; ++ni) acc[mi][ni] = zacc;

    float4 rw[4];      // staged Wd fp32

    auto loadW = [&](int k0) {
        #pragma unroll
        for (int i = 0; i < 4; ++i) {
            const int idx = tid + i * 256;
            const int r   = idx >> 3;
            const int c4  = idx & 7;
            rw[i] = *(const float4*)(Wd + (size_t)(colBase + r) * D_FF + k0 + c4 * 4);
        }
    };
    auto storeW = [&](int p) {
        #pragma unroll
        for (int i = 0; i < 4; ++i) {
            const int idx = tid + i * 256;
            const int r   = idx >> 3;
            const int c4  = idx & 7;
            *(v4bf*)&Bs[p][r * LDAP + c4 * 4] = pack4(rw[i]);
        }
    };

#if !HAVE_TDM
    v8bf rh[2];        // fallback staging for H
    auto loadH = [&](int k0) {
        #pragma unroll
        for (int i = 0; i < 2; ++i) {
            const int c  = tid + i * 256;
            const int r  = c >> 2;
            const int c8 = c & 3;
            rh[i] = *(const v8bf*)(H + (size_t)(rowBase + r) * D_FF + k0 + c8 * 8);
        }
    };
    auto storeH = [&](int p) {
        #pragma unroll
        for (int i = 0; i < 2; ++i) {
            const int c  = tid + i * 256;
            const int r  = c >> 2;
            const int c8 = c & 3;
            *(v8bf*)&As[p][r * LDAP + c8 * 8] = rh[i];
        }
    };
#endif

#if HAVE_TDM
    if (wave == 0) tdm_load_H_tile(H + (size_t)rowBase * D_FF, &As[0][0]);
    loadW(0);
#else
    loadH(0);
    loadW(0);
#endif

    int p = 0;
    for (int k0 = 0; k0 < D_FF; k0 += 32, p ^= 1) {
#if HAVE_TDM
        storeW(p);
        if (wave == 0) __builtin_amdgcn_s_wait_tensorcnt(0);  // As[p] DMA complete
        __syncthreads();                                      // publish As[p], Bs[p]
        if (k0 + 32 < D_FF) {
            if (wave == 0)
                tdm_load_H_tile(H + (size_t)rowBase * D_FF + k0 + 32, &As[p ^ 1][0]);
            loadW(k0 + 32);
        }
#else
        storeH(p);
        storeW(p);
        __syncthreads();
        if (k0 + 32 < D_FF) { loadH(k0 + 32); loadW(k0 + 32); }
#endif

        v16bf aF[2];
        #pragma unroll
        for (int mi = 0; mi < 2; ++mi) {
            const int m = waveM * 32 + mi * 16 + n16;
            const v8bf lo = *(const v8bf*)&As[p][m * LDAP + khalf * 8];
            const v8bf hi = *(const v8bf*)&As[p][m * LDAP + khalf * 8 + 16];
            aF[mi] = cat8(lo, hi);
        }
        #pragma unroll
        for (int ni = 0; ni < 4; ++ni) {
            const int n = waveN * 64 + ni * 16 + n16;
            const v8bf bl = *(const v8bf*)&Bs[p][n * LDAP + khalf * 8];
            const v8bf bh = *(const v8bf*)&Bs[p][n * LDAP + khalf * 8 + 16];
            const v16bf bF = cat8(bl, bh);
            #pragma unroll
            for (int mi = 0; mi < 2; ++mi) {
                acc[mi][ni] = __builtin_amdgcn_wmma_f32_16x16x32_bf16(
                    false, aF[mi], false, bF, (short)0, acc[mi][ni], false, false);
            }
        }
    }

    #pragma unroll
    for (int mi = 0; mi < 2; ++mi) {
        #pragma unroll
        for (int ni = 0; ni < 4; ++ni) {
            #pragma unroll
            for (int r = 0; r < 8; ++r) {
                const int row = rowBase + waveM * 32 + mi * 16 + r + 8 * khalf;
                const int col = colBase + waveN * 64 + ni * 16 + n16;
                out[(size_t)row * D_MODEL + col] = acc[mi][ni][r];
            }
        }
    }
}

// ---------------------------------------------------------------------------
extern "C" void kernel_launch(void* const* d_in, const int* in_sizes, int n_in,
                              void* d_out, int out_size, void* d_ws, size_t ws_size,
                              hipStream_t stream) {
    (void)in_sizes; (void)n_in; (void)out_size; (void)ws_size;
    const float* x  = (const float*)d_in[0];   // [8192, 4096]
    const float* Wg = (const float*)d_in[1];   // [11008, 4096]
    const float* Wu = (const float*)d_in[2];   // [11008, 4096]
    const float* Wd = (const float*)d_in[3];   // [4096, 11008]
    float* out = (float*)d_out;                // [8192, 4096]
    unsigned short* H = (unsigned short*)d_ws; // bf16 [8192, 11008] = ~172 MB scratch

    dim3 blk(256);
    dim3 grdA(TOKENS / 128, D_FF / 128);       // (64, 86)
    fused_gateup_kernel<<<grdA, blk, 0, stream>>>(x, Wg, Wu, H);

    dim3 grdB(TOKENS / 128, D_MODEL / 128);    // (64, 32)
    down_proj_kernel<<<grdB, blk, 0, stream>>>(H, Wd, out);
}
